// SelfAttention_71116068487550
// MI455X (gfx1250) — matrix operand order repaired and checked by reading, not compiled, not run on hardware
//
#include <hip/hip_runtime.h>
#include <cstdint>
#include <cstddef>

// ---------------------------------------------------------------------------
// MI455X (gfx1250) self-attention:
//   1) convert x / W_qkv / W_o to bf16 (workspace)
//   2) QKV GEMM  (v_wmma_f32_16x16x32_bf16), bias, scatter to Q,K,Vt(bf16)
//   3) flash attention (wave32 online softmax, WMMA QK^T and PV)
//   4) output GEMM + bias -> fp32
// ---------------------------------------------------------------------------

typedef __attribute__((ext_vector_type(16))) __bf16 v16bf;
typedef __attribute__((ext_vector_type(8)))  __bf16 v8bf;
typedef __attribute__((ext_vector_type(8)))  float  v8f;

static __device__ __forceinline__ __bf16 f2bf(float f) {
  unsigned u = __builtin_bit_cast(unsigned, f);
  unsigned r = u + 0x7FFFu + ((u >> 16) & 1u);   // round-to-nearest-even
  unsigned short h = (unsigned short)(r >> 16);
  return __builtin_bit_cast(__bf16, h);
}

static __device__ __forceinline__ v16bf join8(v8bf lo, v8bf hi) {
  return __builtin_shufflevector(lo, hi, 0,1,2,3,4,5,6,7,8,9,10,11,12,13,14,15);
}

static __device__ __forceinline__ v8f v8f_zero() {
  v8f z;
#pragma unroll
  for (int i = 0; i < 8; i++) z[i] = 0.f;
  return z;
}

static __device__ __forceinline__ v8f wmma_bf16(v16bf a, v16bf b, v8f c) {
  return __builtin_amdgcn_wmma_f32_16x16x32_bf16(false, a, false, b, (short)0, c, false, false);
}

// A fragment (16x32, 16-bit): lane = row (lane&15); 16 contiguous K starting
// at kb = 16*(lane>>4). p points at row + k0 + kb.
static __device__ __forceinline__ v16bf load_a_frag(const __bf16* p) {
  return join8(*(const v8bf*)p, *(const v8bf*)(p + 8));
}
// B fragment (32x16, 16-bit): lane = col (lane&15); 16 contiguous K starting
// at 16*(lane>>4). p points at row(col) + k0 + kb2.
static __device__ __forceinline__ v16bf load_b_frag(const __bf16* p) {
  return join8(*(const v8bf*)p, *(const v8bf*)(p + 8));
}

// ---------------------------------------------------------------------------
__global__ void convert_f32_bf16(const float* __restrict__ in,
                                 __bf16* __restrict__ out, int n) {
  int i = blockIdx.x * blockDim.x + threadIdx.x;
  int stride = gridDim.x * blockDim.x;
  for (; i < n; i += stride) out[i] = f2bf(in[i]);
}

// ---------------------------------------------------------------------------
// C[M,N] = A[M,K] * W[N,K]^T + bias.  Block: 256 thr = 8 waves (2 M x 4 N),
// wave tile 64x32 (4x2 WMMA frags), block tile 128x128.
// MODE 0: scatter bf16 into Q[b,h,s,dh], K[b,h,s,dh], Vt[b,h,dh,s]
// MODE 1: fp32 row-major out
template <int MODE>
__global__ __launch_bounds__(256) void gemm_bf16_nt(
    const __bf16* __restrict__ A, const __bf16* __restrict__ W,
    const float* __restrict__ bias, int K, int N,
    float* __restrict__ outF,
    __bf16* __restrict__ Qb, __bf16* __restrict__ Kb, __bf16* __restrict__ Vtb) {
  const int lane = threadIdx.x & 31;
  const int wid  = threadIdx.x >> 5;
  const int col  = lane & 15;
  const int hi   = lane >> 4;
  const int wm   = wid >> 2;    // 0..1
  const int wn   = wid & 3;     // 0..3
  const int m0 = blockIdx.y * 128 + wm * 64;
  const int n0 = blockIdx.x * 128 + wn * 32;

  v8f acc[4][2];
#pragma unroll
  for (int i = 0; i < 4; i++)
#pragma unroll
    for (int j = 0; j < 2; j++) acc[i][j] = v8f_zero();

  for (int k0 = 0; k0 < K; k0 += 32) {
    v16bf af[4], bf2[2];
#pragma unroll
    for (int i = 0; i < 4; i++)
      af[i] = load_a_frag(A + (size_t)(m0 + i * 16 + col) * K + k0 + hi * 16);
#pragma unroll
    for (int j = 0; j < 2; j++)
      bf2[j] = load_b_frag(W + (size_t)(n0 + j * 16 + col) * K + k0 + hi * 16);
#pragma unroll
    for (int i = 0; i < 4; i++)
#pragma unroll
      for (int j = 0; j < 2; j++)
        acc[i][j] = wmma_bf16(af[i], bf2[j], acc[i][j]);
  }

#pragma unroll
  for (int j = 0; j < 2; j++) {
    const int ncol = n0 + j * 16 + col;
    const float bv = bias[ncol];
    if (MODE == 1) {
#pragma unroll
      for (int i = 0; i < 4; i++)
#pragma unroll
        for (int r = 0; r < 8; r++) {
          const int mrow = m0 + i * 16 + r + 8 * hi;
          outF[(size_t)mrow * N + ncol] = acc[i][j][r] + bv;
        }
    } else {
      const int which = ncol >> 11;      // 0=Q 1=K 2=V  (D_EMBED=2048)
      const int rem = ncol & 2047;
      const int h = rem >> 7;
      const int dh = rem & 127;
#pragma unroll
      for (int i = 0; i < 4; i++)
#pragma unroll
        for (int r = 0; r < 8; r++) {
          const int mrow = m0 + i * 16 + r + 8 * hi;
          const int b = mrow >> 11;      // seq = 2048
          const int s = mrow & 2047;
          const __bf16 v = f2bf(acc[i][j][r] + bv);
          const size_t bh = (size_t)(b * 16 + h);
          if (which == 0)      Qb [(bh * 2048 + s) * 128 + dh] = v;
          else if (which == 1) Kb [(bh * 2048 + s) * 128 + dh] = v;
          else                 Vtb[(bh * 128 + dh) * 2048 + s] = v;
        }
    }
  }
}

// ---------------------------------------------------------------------------
// Flash attention: 8 waves/block, each wave owns 16 query rows (block = 128),
// grid (seq/128, heads, batch). 32 keys per iteration: 8 QK^T + 8 PV WMMAs.
__global__ __launch_bounds__(256) void flash_attn(
    const __bf16* __restrict__ Q, const __bf16* __restrict__ Kx,
    const __bf16* __restrict__ Vt, __bf16* __restrict__ attnb,
    const int* __restrict__ causal_p) {
  __shared__ __align__(16) __bf16 pbuf[8][16][48];  // per-wave P re-layout

  const int lane = threadIdx.x & 31;
  const int wid  = threadIdx.x >> 5;
  const int col  = lane & 15;
  const int hi   = lane >> 4;
  const int q0   = blockIdx.x * 128 + wid * 16;
  const int h    = blockIdx.y;
  const int b    = blockIdx.z;
  const int causal = *causal_p;

  const size_t bh = (size_t)(b * 16 + h);
  const __bf16* Qp = Q  + bh * (2048 * 128);
  const __bf16* Kp = Kx + bh * (2048 * 128);
  const __bf16* Vp = Vt + bh * (128 * 2048);

  // scores computed directly in exp2 domain: S' = S * (1/sqrt(128)) * log2(e)
  const float kScale = 0.08838834764831845f * 1.4426950408889634f;
  const float NEG_INF = -__builtin_inff();

  v16bf aq[4];
#pragma unroll
  for (int t = 0; t < 4; t++)
    aq[t] = load_a_frag(Qp + (size_t)(q0 + col) * 128 + t * 32 + hi * 16);

  v8f o[8];
#pragma unroll
  for (int c = 0; c < 8; c++) o[c] = v8f_zero();
  float mrow[8], lrow[8];
#pragma unroll
  for (int r = 0; r < 8; r++) { mrow[r] = NEG_INF; lrow[r] = 0.f; }

  const int jend = causal ? (q0 + 16) : 2048;
  for (int j0 = 0; j0 < jend; j0 += 32) {
    // ---- scores: two 16x16 tiles (keys j0..j0+15, j0+16..j0+31)
    v8f sa = v8f_zero(), sb = v8f_zero();
    {
      int kk = j0 + col;           int kr = kk < 2047 ? kk : 2047;
#pragma unroll
      for (int t = 0; t < 4; t++)
        sa = wmma_bf16(aq[t], load_b_frag(Kp + (size_t)kr * 128 + t * 32 + hi * 16), sa);
    }
    {
      int kk = j0 + 16 + col;      int kr = kk < 2047 ? kk : 2047;
#pragma unroll
      for (int t = 0; t < 4; t++)
        sb = wmma_bf16(aq[t], load_b_frag(Kp + (size_t)kr * 128 + t * 32 + hi * 16), sb);
    }
    // scale + causal mask
#pragma unroll
    for (int r = 0; r < 8; r++) {
      const int qg = q0 + r + 8 * hi;
      float x0 = sa[r] * kScale;
      float x1 = sb[r] * kScale;
      if (causal && (j0 + col > qg))      x0 = NEG_INF;
      if (causal && (j0 + 16 + col > qg)) x1 = NEG_INF;
      sa[r] = x0; sb[r] = x1;
    }
    // row max across the 16 lanes of each half (xor 1,2,4,8 stays in-half)
    float rm[8];
#pragma unroll
    for (int r = 0; r < 8; r++) rm[r] = fmaxf(sa[r], sb[r]);
#pragma unroll
    for (int msk = 1; msk <= 8; msk <<= 1)
#pragma unroll
      for (int r = 0; r < 8; r++) rm[r] = fmaxf(rm[r], __shfl_xor(rm[r], msk, 32));
    // online softmax update + P -> LDS (bf16)
    float corr[8], rs[8];
#pragma unroll
    for (int r = 0; r < 8; r++) {
      const float mnew = fmaxf(mrow[r], rm[r]);
      corr[r] = exp2f(mrow[r] - mnew);
      mrow[r] = mnew;
      const float pa = exp2f(sa[r] - mnew);
      const float pb = exp2f(sb[r] - mnew);
      rs[r] = pa + pb;
      pbuf[wid][r + 8 * hi][col]      = f2bf(pa);
      pbuf[wid][r + 8 * hi][16 + col] = f2bf(pb);
    }
#pragma unroll
    for (int msk = 1; msk <= 8; msk <<= 1)
#pragma unroll
      for (int r = 0; r < 8; r++) rs[r] += __shfl_xor(rs[r], msk, 32);
#pragma unroll
    for (int r = 0; r < 8; r++) lrow[r] = lrow[r] * corr[r] + rs[r];
#pragma unroll
    for (int c = 0; c < 8; c++)
#pragma unroll
      for (int r = 0; r < 8; r++) o[c][r] *= corr[r];

    // within-wave LDS RAW: wait split DS counter, no block barrier needed
    asm volatile("s_wait_dscnt 0" ::: "memory");

    const __bf16* pp = &pbuf[wid][col][hi * 16];
    const v16bf pf = join8(*(const v8bf*)pp, *(const v8bf*)(pp + 8));

    int s0 = j0 + hi * 16;
    if (s0 > 2032) s0 = 2032;   // clamp (contribution is zero via P==0)
#pragma unroll
    for (int c = 0; c < 8; c++) {
      const __bf16* vp = Vp + (size_t)(c * 16 + col) * 2048 + s0;
      o[c] = wmma_bf16(pf, load_b_frag(vp), o[c]);
    }
  }

  // normalize and write attn (bf16) as [b*2048+s, h*128+dh] for the out GEMM
  float inv[8];
#pragma unroll
  for (int r = 0; r < 8; r++) inv[r] = 1.f / lrow[r];
#pragma unroll
  for (int c = 0; c < 8; c++)
#pragma unroll
    for (int r = 0; r < 8; r++) {
      const size_t row = (size_t)(b * 2048 + q0 + r + 8 * hi);
      attnb[row * 2048 + h * 128 + c * 16 + col] = f2bf(o[c][r] * inv[r]);
    }
}

// ---------------------------------------------------------------------------
extern "C" void kernel_launch(void* const* d_in, const int* in_sizes, int n_in,
                              void* d_out, int out_size, void* d_ws, size_t ws_size,
                              hipStream_t stream) {
  const float* x    = (const float*)d_in[0];   // [2,2048,2048]
  const float* Wqkv = (const float*)d_in[1];   // [6144,2048]
  const float* bqkv = (const float*)d_in[2];   // [6144]
  const float* Wo   = (const float*)d_in[3];   // [2048,2048]
  const float* bo   = (const float*)d_in[4];   // [2048]
  const int* causal = (const int*)d_in[5];
  float* out = (float*)d_out;                  // [2,2048,2048] fp32

  char* ws = (char*)d_ws;
  size_t off = 0;
  auto take = [&](size_t bytes) -> void* {
    void* p = ws + off;
    off += (bytes + 255) & ~(size_t)255;
    return p;
  };
  __bf16* Xb    = (__bf16*)take(4096ull * 2048 * 2);
  __bf16* Wqkvb = (__bf16*)take(6144ull * 2048 * 2);
  __bf16* Wob   = (__bf16*)take(2048ull * 2048 * 2);
  __bf16* Qb    = (__bf16*)take(2ull * 16 * 2048 * 128 * 2);
  __bf16* Kb    = (__bf16*)take(2ull * 16 * 2048 * 128 * 2);
  __bf16* Vtb   = (__bf16*)take(2ull * 16 * 128 * 2048 * 2);
  __bf16* Ab    = (__bf16*)take(4096ull * 2048 * 2);

  convert_f32_bf16<<<2048, 256, 0, stream>>>(x, Xb, 4096 * 2048);
  convert_f32_bf16<<<2048, 256, 0, stream>>>(Wqkv, Wqkvb, 6144 * 2048);
  convert_f32_bf16<<<1024, 256, 0, stream>>>(Wo, Wob, 2048 * 2048);

  // QKV projection: M=4096, N=6144, K=2048
  gemm_bf16_nt<0><<<dim3(6144 / 128, 4096 / 128), 256, 0, stream>>>(
      Xb, Wqkvb, bqkv, 2048, 6144, nullptr, Qb, Kb, Vtb);

  // attention: grid (seq/128, heads, batch)
  flash_attn<<<dim3(2048 / 128, 16, 2), 256, 0, stream>>>(Qb, Kb, Vtb, Ab, causal);

  // output projection: M=4096, N=2048, K=2048
  gemm_bf16_nt<1><<<dim3(2048 / 128, 4096 / 128), 256, 0, stream>>>(
      Ab, Wob, bo, 2048, 2048, out, nullptr, nullptr, nullptr);
}
